// NonUQEntropyModel_33921651703829
// MI455X (gfx1250) — matrix operand order repaired and checked by reading, not compiled, not run on hardware
//
#include <hip/hip_runtime.h>
#include <hip/hip_bf16.h>

// ---------------------------------------------------------------------------
// MI455X (gfx1250) fused NonUQEntropyModel forward.
// 24->24 masked 3x3x3 convs as bf16 WMMA GEMMs (f32 accum).
// K padded per-tap to 32 (K' = tap*32 + c, c<24 real), 14 chunks of 32.
// Staged LDS tile is channel-padded to 32 (c 24..31 zeroed), so every
// A-fragment is two hazard-free ds_load_b128 at immediate offsets.
// ---------------------------------------------------------------------------

typedef __attribute__((ext_vector_type(16))) __bf16          v16bf;
typedef __attribute__((ext_vector_type(8)))  float           v8f;
typedef __attribute__((ext_vector_type(8)))  unsigned short  u16x8;
typedef __attribute__((ext_vector_type(16))) unsigned short  u16x16;

#define NB   4
#define DD   16    // depth (= channel dim of x)
#define HH   128
#define WW   128
#define KCH  24
#define CP   32    // channel-padded stride in staged tile
#define LQ   6
#define NTAP 14    // mask-B taps 0..13 (center included)
// per-conv weight fragments: 2 tiles x 14 chunks x 512 halves
#define WBHALVES (2 * NTAP * 512)

static __device__ __forceinline__ unsigned short f2bf(float f) {
  unsigned u = __float_as_uint(f);
  u += 0x7fffu + ((u >> 16) & 1u);          // round-to-nearest-even
  return (unsigned short)(u >> 16);
}
static __device__ __forceinline__ float bf2f(unsigned short h) {
  return __uint_as_float(((unsigned)h) << 16);
}

// ---------------------------------------------------------------- pad fill --
__global__ void pad_fill_k(unsigned short* __restrict__ v,
                           const float* __restrict__ centers, int n) {
  int i = blockIdx.x * blockDim.x + threadIdx.x;
  if (i < n) v[i] = f2bf(centers[0]);
}

// ---------------------------------------------------------------- quantize --
// qbar == qhard in forward (STE). Writes qbar (fp32 NHWC), padded bf16 volume
// v[n][18][130][130], symbols u8 [n][16][128][128].
__global__ void quantize_k(const float* __restrict__ x,
                           const float* __restrict__ centers,
                           float* __restrict__ qbar,
                           unsigned short* __restrict__ v,
                           unsigned char* __restrict__ sym) {
  int i = blockIdx.x * blockDim.x + threadIdx.x;   // exact grid
  float xv = x[i];
  int best = 0; float bd = 3.4e38f;
#pragma unroll
  for (int l = 0; l < LQ; ++l) {
    float d = xv - centers[l]; d *= d;
    if (d < bd) { bd = d; best = l; }
  }
  float qh = centers[best];
  qbar[i] = qh;
  int c = i & 15, w = (i >> 4) & 127, h = (i >> 11) & 127, n = i >> 18;
  v[(((n * 18) + (c + 1)) * 130 + (h + 1)) * 130 + (w + 1)] = f2bf(qh);
  sym[(((n * DD) + c) * HH + h) * WW + w] = (unsigned char)best;
}

// ------------------------------------------------------------ pack weights --
// bf16 B-fragment layout (16-bit B 32x16): lane&15 = N column;
// lanes 0-15 hold K=0..7,16..23; lanes 16-31 hold K=8..15,24..31 of each chunk.
// K' = chunk*32 + c, chunk = tap (0..13), c<24 real else 0.
__global__ void pack_w_k(const float* __restrict__ w1,
                         const float* __restrict__ w2,
                         unsigned short* __restrict__ wB) {
  const int TOT = 2 * WBHALVES;
  for (int i = blockIdx.x * blockDim.x + threadIdx.x; i < TOT;
       i += gridDim.x * blockDim.x) {
    int conv = i / WBHALVES;
    int r    = i - conv * WBHALVES;
    int tile = r / (NTAP * 512);
    int r2   = r - tile * (NTAP * 512);
    int t    = r2 >> 9;            // chunk == tap
    int q    = r2 & 511;
    int lane = q >> 4;
    int e    = q & 15;
    int c  = ((e < 8) ? 0 : 16) + ((lane < 16) ? 0 : 8) + (e & 7);
    int Nn = tile * 16 + (lane & 15);
    float val = 0.f;
    if (c < KCH && Nn < KCH)
      val = (conv ? w2 : w1)[(Nn * KCH + c) * 27 + t];
    wB[i] = f2bf(val);
  }
}

// -------------------------------------------------------------------- conv0 --
// mask-A (taps 0..12), 1 -> 24 ch, relu, bf16 channels-last store.
__global__ __launch_bounds__(256) void conv0_k(const unsigned short* __restrict__ v,
                                               const float* __restrict__ w0,
                                               const float* __restrict__ b0,
                                               unsigned short* __restrict__ h0) {
  int vi = blockIdx.x * 256 + threadIdx.x;         // 16*128*128 exact
  int z = vi >> 14, y = (vi >> 7) & 127, x = vi & 127;
  float val[13];
#pragma unroll
  for (int t = 0; t < 13; ++t) {
    int dz = t / 9, rem = t - dz * 9, dy = rem / 3, dx = rem - dy * 3;
    val[t] = bf2f(v[((z + dz) * 130 + (y + dy)) * 130 + (x + dx)]);
  }
  unsigned* out = (unsigned*)(h0 + (size_t)vi * KCH);
#pragma unroll
  for (int k = 0; k < KCH; k += 2) {
    float a0 = b0[k], a1 = b0[k + 1];
#pragma unroll
    for (int t = 0; t < 13; ++t) {
      a0 = fmaf(w0[k * 27 + t], val[t], a0);
      a1 = fmaf(w0[(k + 1) * 27 + t], val[t], a1);
    }
    a0 = fmaxf(a0, 0.f); a1 = fmaxf(a1, 0.f);
    out[k >> 1] = (unsigned)f2bf(a0) | ((unsigned)f2bf(a1) << 16);
  }
}

// ---------------------------------------------------- WMMA strip (shared) ----
// Per wave: 16-voxel x-strip at (z,y). Stage 9 rows x 18 x x 32c (padded)
// bf16 in LDS; run 14 taps x 2 N-tiles of v_wmma_f32_16x16x32_bf16.
#define STAGE_HALVES (162 * CP)   // 5184 halves (10368 B) per wave
#define NWAVE 4                   // waves per block (128 threads)

__device__ __forceinline__ void wmma_strip(const unsigned short* __restrict__ hin,
                                           const unsigned short* __restrict__ wB,
                                           unsigned short* Sw,
                                           int z, int y, int x0, int lane,
                                           v8f& acc0, v8f& acc1) {
  // ---- halo stage: 162 sites (row=dz*3+dy, xx); parts 0..2 real, part 3 zero
  for (int i = lane; i < 162; i += 32) {
    unsigned row = (unsigned)i / 18u;              // 0..8
    int xx = i - (int)row * 18;
    int dz = (int)row / 3, dy = (int)row - dz * 3;
    int zz = z + dz - 1, yy = y + dy - 1, xg = x0 + xx - 1;
    bool ok = (unsigned)zz < (unsigned)DD && (unsigned)yy < (unsigned)HH &&
              (unsigned)xg < (unsigned)WW;
    const u16x8* src =
        (const u16x8*)(hin + (size_t)((zz * HH + yy) * WW + xg) * KCH);
#pragma unroll
    for (int part = 0; part < 4; ++part) {
      u16x8 val = {0, 0, 0, 0, 0, 0, 0, 0};
      if (part < 3 && ok) val = src[part];
      *(u16x8*)(Sw + i * CP + part * 8) = val;
    }
  }
  asm volatile("s_wait_dscnt 0" ::: "memory");

  const v16bf* wBv = (const v16bf*)wB;
  int m = lane & 15;                               // A row (voxel)
  bool lo16 = lane < 16;
  // lanes 0-15: e0-7 -> c=0..7,  e8-15 -> c=16..23
  // lanes16-31: e0-7 -> c=8..15, e8-15 -> c=24..31 (LDS-zeroed)
  const unsigned short* pA0 = Sw + m * CP + (lo16 ? 0 : 8);
  const unsigned short* pA1 = Sw + m * CP + (lo16 ? 16 : 24);
#pragma unroll
  for (int t = 0; t < NTAP; ++t) {
    const int row = t / 3;                         // dz*3+dy (compile-time)
    const int dxv = t - row * 3;
    const int off = (row * 18 + dxv) * CP;         // immediate LDS offset
    u16x8 l0 = *(const u16x8*)(pA0 + off);
    u16x8 l1 = *(const u16x8*)(pA1 + off);
    u16x16 cat = __builtin_shufflevector(l0, l1, 0, 1, 2, 3, 4, 5, 6, 7,
                                         8, 9, 10, 11, 12, 13, 14, 15);
    v16bf a = __builtin_bit_cast(v16bf, cat);
    v16bf bv0 = wBv[(0 * NTAP + t) * 32 + lane];
    v16bf bv1 = wBv[(1 * NTAP + t) * 32 + lane];
    acc0 = __builtin_amdgcn_wmma_f32_16x16x32_bf16(false, a, false, bv0,
                                                   (short)0, acc0, false, false);
    acc1 = __builtin_amdgcn_wmma_f32_16x16x32_bf16(false, a, false, bv1,
                                                   (short)0, acc1, false, false);
  }
}

// -------------------------------------------------------------------- conv1 --
// grid: DD*HH*2 blocks of 128 threads; each block = one 64-voxel x half-row.
__global__ __launch_bounds__(128) void conv1_k(const unsigned short* __restrict__ hin,
                                               const unsigned short* __restrict__ wB,
                                               const float* __restrict__ bias,
                                               unsigned short* __restrict__ hout) {
  __shared__ unsigned short S[NWAVE * STAGE_HALVES];
  int half = blockIdx.x & 1;
  int zy = blockIdx.x >> 1; int z = zy >> 7; int y = zy & 127;
  int wv = threadIdx.x >> 5; int lane = threadIdx.x & 31;
  int x0 = half * 64 + (wv << 4);
  unsigned short* Sw = S + wv * STAGE_HALVES;

  v8f acc0 = {0, 0, 0, 0, 0, 0, 0, 0}; v8f acc1 = acc0;
  wmma_strip(hin, wB, Sw, z, y, x0, lane, acc0, acc1);

  int n0 = lane & 15;
  bool n1ok = n0 < 8;
  float bb0 = bias[n0];
  float bb1 = n1ok ? bias[16 + n0] : 0.f;
  int mofs = (lane < 16) ? 0 : 8;                  // D layout: VGPR r -> M=r(+8)
#pragma unroll
  for (int r = 0; r < 8; ++r) {
    size_t voxel = (size_t)((z * HH + y) * WW + x0 + r + mofs) * KCH;
    hout[voxel + n0] = f2bf(fmaxf(acc0[r] + bb0, 0.f));
    if (n1ok) hout[voxel + 16 + n0] = f2bf(fmaxf(acc1[r] + bb1, 0.f));
  }
}

// ------------------------------------------------------- conv2 + head fused --
__global__ __launch_bounds__(128) void conv2_k(const unsigned short* __restrict__ h1,
                                               const unsigned short* __restrict__ wB,
                                               const float* __restrict__ b2,
                                               const unsigned short* __restrict__ h0,
                                               const float* __restrict__ w_out,
                                               const float* __restrict__ b_out,
                                               const unsigned char* __restrict__ sym,
                                               float* __restrict__ out) {
  __shared__ unsigned short S[NWAVE * STAGE_HALVES];
  int half = blockIdx.x & 1;
  int zy = blockIdx.x >> 1; int z = zy >> 7; int y = zy & 127;
  int wv = threadIdx.x >> 5; int lane = threadIdx.x & 31;
  int x0 = half * 64 + (wv << 4);
  unsigned short* Sw = S + wv * STAGE_HALVES;

  v8f acc0 = {0, 0, 0, 0, 0, 0, 0, 0}; v8f acc1 = acc0;
  wmma_strip(h1, wB, Sw, z, y, x0, lane, acc0, acc1);
  asm volatile("s_wait_dscnt 0" ::: "memory");

  // residual + relu -> per-wave LDS transpose buffer (reuse staging area)
  float* Sf = (float*)Sw;                          // 16 voxels * 24 ch fp32
  int n0 = lane & 15;
  bool n1ok = n0 < 8;
  int mofs = (lane < 16) ? 0 : 8;
#pragma unroll
  for (int r = 0; r < 8; ++r) {
    int vloc = r + mofs;
    size_t voxel = (size_t)((z * HH + y) * WW + x0 + vloc) * KCH;
    float t0 = acc0[r] + b2[n0] + bf2f(h0[voxel + n0]);
    Sf[vloc * KCH + n0] = fmaxf(t0, 0.f);
    if (n1ok) {
      float t1 = acc1[r] + b2[16 + n0] + bf2f(h0[voxel + 16 + n0]);
      Sf[vloc * KCH + 16 + n0] = fmaxf(t1, 0.f);
    }
  }
  asm volatile("s_wait_dscnt 0" ::: "memory");

  // 24->6 head + log_softmax + bitcost, one voxel per lane (lanes 0..15)
  if (lane < 16) {
    int gx = x0 + lane;
    float rv[KCH];
#pragma unroll
    for (int c = 0; c < KCH; ++c) rv[c] = Sf[lane * KCH + c];
    float logit[LQ];
#pragma unroll
    for (int l = 0; l < LQ; ++l) {
      float a = b_out[l];
#pragma unroll
      for (int c = 0; c < KCH; ++c) a = fmaf(w_out[l * KCH + c], rv[c], a);
      logit[l] = a;
    }
    float mx = logit[0];
#pragma unroll
    for (int l = 1; l < LQ; ++l) mx = fmaxf(mx, logit[l]);
    float s = 0.f;
#pragma unroll
    for (int l = 0; l < LQ; ++l) s += __expf(logit[l] - mx);
    float lse = mx + __logf(s);
    int sv = sym[(z * HH + y) * WW + gx];
    float bc = (lse - logit[sv]) * 1.4426950408889634f;  // /ln2
    out[(y * WW + gx) * DD + z] = bc;                    // NHWC (c = depth z)
  }
}

// ------------------------------------------------------------------ launch --
extern "C" void kernel_launch(void* const* d_in, const int* in_sizes, int n_in,
                              void* d_out, int out_size, void* d_ws, size_t ws_size,
                              hipStream_t stream) {
  const float* x       = (const float*)d_in[0];
  const float* centers = (const float*)d_in[1];
  const float* w0      = (const float*)d_in[2];
  const float* b0      = (const float*)d_in[3];
  const float* w1      = (const float*)d_in[4];
  const float* b1      = (const float*)d_in[5];
  const float* w2      = (const float*)d_in[6];
  const float* b2      = (const float*)d_in[7];
  const float* w_out   = (const float*)d_in[8];
  const float* b_out   = (const float*)d_in[9];
  float* out = (float*)d_out;

  char* ws = (char*)d_ws;
  size_t off = 0;
  auto carve = [&](size_t bytes) -> void* {
    void* p = ws + off;
    off = (off + bytes + 255) & ~(size_t)255;
    return p;
  };
  const size_t VPLANE = (size_t)18 * 130 * 130;            // padded volume per n
  unsigned short* v   = (unsigned short*)carve(2 * NB * VPLANE);
  unsigned char*  sym = (unsigned char*)carve((size_t)NB * DD * HH * WW);
  unsigned short* wB  = (unsigned short*)carve(2ull * 2 * WBHALVES);
  const size_t HVOX = (size_t)DD * HH * WW;                // 262144 voxels per n
  unsigned short* h0  = (unsigned short*)carve(2 * HVOX * KCH);
  unsigned short* h1  = (unsigned short*)carve(2 * HVOX * KCH);

  const int VTOT = (int)(NB * VPLANE);
  pad_fill_k<<<(VTOT + 255) / 256, 256, 0, stream>>>(v, centers, VTOT);
  quantize_k<<<(NB * HH * WW * DD) / 256, 256, 0, stream>>>(x, centers, out, v, sym);
  pack_w_k<<<8, 256, 0, stream>>>(w1, w2, wB);

  float* out_bc = out + (size_t)NB * HH * WW * DD;
  for (int n = 0; n < NB; ++n) {
    conv0_k<<<(int)(HVOX / 256), 256, 0, stream>>>(v + n * VPLANE, w0, b0, h0);
    conv1_k<<<DD * HH * 2, 128, 0, stream>>>(h0, wB, b1, h1);
    conv2_k<<<DD * HH * 2, 128, 0, stream>>>(h1, wB + WBHALVES, b2, h0,
                                             w_out, b_out, sym + n * HVOX,
                                             out_bc + n * (size_t)HH * WW * DD);
  }
}
